// CutCrossEntropyLoss_30769145708793
// MI455X (gfx1250) — matrix operand order repaired and checked by reading, not compiled, not run on hardware
//
#include <hip/hip_runtime.h>

typedef __bf16 bf16_t;
typedef bf16_t   v16bf  __attribute__((ext_vector_type(16)));
typedef bf16_t   v4bf   __attribute__((ext_vector_type(4)));
typedef float    v8f    __attribute__((ext_vector_type(8)));
typedef float    f32x4  __attribute__((ext_vector_type(4)));
typedef unsigned u32x4  __attribute__((ext_vector_type(4)));

#define IGNORE_INDEX (-100)

namespace {
constexpr int kB = 2;
constexpr int kS = 2048;
constexpr int kD = 2048;
constexpr int kV = 50257;
constexpr int kM = kB * (kS - 1);          // 4094 token rows
constexpr int kMP = 4096;                  // padded rows in workspace
constexpr int MT = 256, NT = 256, KT = 64; // block tile
constexpr int NTILES = (kV + NT - 1) / NT; // 197
constexpr int NCH = NTILES * 4;            // 788 softmax chunks (64 cols each)
constexpr int PITCH = 72;                  // LDS row pitch in bf16 (64 + pad, 144B)
constexpr float LOG2E = 1.4426950408889634f;
constexpr float LN2   = 0.6931471805599453f;
}

// round-to-nearest-even f32 -> bf16 (pure integer ops; co-executes with WMMA)
__device__ __forceinline__ bf16_t f2bf(float f) {
  unsigned u = __builtin_bit_cast(unsigned, f);
  unsigned r = (u + 0x7FFFu + ((u >> 16) & 1u)) >> 16;
  unsigned short s = (unsigned short)r;
  return __builtin_bit_cast(bf16_t, s);
}

union FragU { struct { u32x4 a, b; } q; v16bf v; };

// 16-bit A fragment 16x32 (MxK): lane l<16 -> M=l, K = half*8 + {0..7, 16..23}
__device__ __forceinline__ v16bf load_fragA(const bf16_t* p, int row0, int kk, int lane) {
  int half = lane >> 4, l = lane & 15;
  const bf16_t* q = p + (row0 + l) * PITCH + kk * 32 + half * 8;
  FragU x;
  x.q.a = *(const u32x4*)(q);       // K = half*8 + 0..7
  x.q.b = *(const u32x4*)(q + 16);  // K = half*8 + 16..23
  return x.v;
}

// 16-bit B fragment 32x16 (KxN): lane l<16 -> N=l, K = half*8 + {0..7, 16..23}
__device__ __forceinline__ v16bf load_fragB(const bf16_t* p, int row0, int kk, int lane) {
  int half = lane >> 4, l = lane & 15;
  const bf16_t* q = p + (row0 + l) * PITCH + kk * 32 + half * 8;
  FragU x;
  x.q.a = *(const u32x4*)(q);
  x.q.b = *(const u32x4*)(q + 16);
  return x.v;
}

// ---------------------------------------------------------------------------
// Kernel 1: fused bf16 WMMA GEMM (logits = e @ W^T + bias) with online-softmax
// partial (max, sum(exp)) per (row, 64-column chunk) written to workspace.
// Software-pipelined K loop: next tile's global loads are issued before the
// WMMA phase (in flight across it); convert + ds_store after the consume-sync.
// Grid: (ceil(M/256), ceil(V/256)) x 512 threads (16 waves, 4x4 wave grid).
// ---------------------------------------------------------------------------
__global__ __launch_bounds__(512, 1)
void fused_logits_lse_kernel(const float* __restrict__ e,
                             const float* __restrict__ w,
                             const float* __restrict__ bias,
                             float* __restrict__ pmax,
                             float* __restrict__ psum) {
  extern __shared__ __align__(16) bf16_t smem[];
  bf16_t* lA = smem;               // MT x PITCH bf16
  bf16_t* lB = smem + MT * PITCH;  // NT x PITCH bf16

  const int tid  = threadIdx.x;
  const int lane = tid & 31;
  const int wave = tid >> 5;
  const int wm = wave >> 2;   // 0..3 (64-row band)
  const int wn = wave & 3;    // 0..3 (64-col band)
  const int l16  = lane & 15;
  const int half = lane >> 4;

  const int rowBlk = blockIdx.x * MT;
  const int colBlk = blockIdx.y * NT;

  const int lc = tid & 15;    // float4 index within a 64-wide K row
  const int lr = tid >> 4;    // 0..31 row group

  // --- hoisted per-thread addressing: 32-bit element offsets + validity ---
  unsigned offA[8], offB[8];
  bool vA[8], vB[8];
  #pragma unroll
  for (int p = 0; p < 8; ++p) {
    int grow = rowBlk + p * 32 + lr;
    vA[p] = (grow < kM);
    int bb = (grow >= (kS - 1)) ? 1 : 0;
    int ss = grow - bb * (kS - 1);
    offA[p] = (unsigned)((bb * kS + ss) * kD + lc * 4);
    int v = colBlk + p * 32 + lr;
    vB[p] = (v < kV);
    offB[p] = (unsigned)(v < kV ? v : 0) * (unsigned)kD + (unsigned)(lc * 4);
  }

  v8f acc[4][4];
  #pragma unroll
  for (int i = 0; i < 4; ++i)
    #pragma unroll
    for (int j = 0; j < 4; ++j)
      acc[i][j] = (v8f)0.0f;

  f32x4 sa[8], sb[8];

  // helper lambdas keep staging in registers
  auto stage_loads = [&](int k0) {
    #pragma unroll
    for (int p = 0; p < 8; ++p) {
      f32x4 f = (f32x4)0.0f;
      if (vA[p]) f = *(const f32x4*)(e + (size_t)offA[p] + (size_t)k0);
      sa[p] = f;
    }
    #pragma unroll
    for (int p = 0; p < 8; ++p) {
      f32x4 f = (f32x4)0.0f;
      if (vB[p]) {
        const float* ptr = w + (size_t)offB[p] + (size_t)k0;
        f = *(const f32x4*)ptr;
        // GL2 prefetch of the tile one K-step ahead of this one (i.e. two
        // ahead of the tile currently being computed). Same address regs,
        // +256B folds into the prefetch immediate; speculative at the edges.
        __builtin_prefetch(ptr + KT);
      }
      sb[p] = f;
    }
  };
  auto publish_lds = [&]() {
    #pragma unroll
    for (int p = 0; p < 8; ++p) {
      int r = p * 32 + lr;
      v4bf pk;
      pk.x = f2bf(sa[p].x); pk.y = f2bf(sa[p].y); pk.z = f2bf(sa[p].z); pk.w = f2bf(sa[p].w);
      *(v4bf*)(lA + r * PITCH + lc * 4) = pk;
    }
    #pragma unroll
    for (int p = 0; p < 8; ++p) {
      int r = p * 32 + lr;
      v4bf pk;
      pk.x = f2bf(sb[p].x); pk.y = f2bf(sb[p].y); pk.z = f2bf(sb[p].z); pk.w = f2bf(sb[p].w);
      *(v4bf*)(lB + r * PITCH + lc * 4) = pk;
    }
  };

  stage_loads(0);
  publish_lds();
  __syncthreads();

  for (int k0 = 0; k0 < kD; k0 += KT) {
    const bool haveNext = (k0 + KT) < kD;
    if (haveNext) stage_loads(k0 + KT);   // loads stay in flight across WMMAs

    // ---- 2 x (K=32) WMMA steps: 32 v_wmma per k0 per wave ----
    #pragma unroll
    for (int kk = 0; kk < 2; ++kk) {
      v16bf bfr[4];
      #pragma unroll
      for (int nf = 0; nf < 4; ++nf)
        bfr[nf] = load_fragB(lB, wn * 64 + nf * 16, kk, lane);
      #pragma unroll
      for (int mf = 0; mf < 4; ++mf) {
        v16bf af = load_fragA(lA, wm * 64 + mf * 16, kk, lane);
        #pragma unroll
        for (int nf = 0; nf < 4; ++nf)
          acc[mf][nf] = __builtin_amdgcn_wmma_f32_16x16x32_bf16(
              false, af, false, bfr[nf], (short)0, acc[mf][nf], false, false);
      }
    }

    __syncthreads();               // everyone done reading current LDS tile
    if (haveNext) publish_lds();   // convert + publish next tile
    __syncthreads();               // next tile visible to all waves
  }

  // ---- epilogue: bias + per-row (over this wave's 64 cols) max & sum(exp) ----
  float bv[4];
  bool  cvalid[4];
  #pragma unroll
  for (int nf = 0; nf < 4; ++nf) {
    int col = colBlk + wn * 64 + nf * 16 + l16;
    cvalid[nf] = (col < kV);
    bv[nf] = cvalid[nf] ? bias[col] : 0.0f;
  }
  const int ch = blockIdx.y * 4 + wn;
  #pragma unroll
  for (int mf = 0; mf < 4; ++mf) {
    #pragma unroll
    for (int j = 0; j < 8; ++j) {
      // C/D layout: VGPR j, lanes 0-15 -> row mf*16+j, lanes 16-31 -> row mf*16+j+8
      float x0 = cvalid[0] ? acc[mf][0][j] + bv[0] : -__builtin_inff();
      float x1 = cvalid[1] ? acc[mf][1][j] + bv[1] : -__builtin_inff();
      float x2 = cvalid[2] ? acc[mf][2][j] + bv[2] : -__builtin_inff();
      float x3 = cvalid[3] ? acc[mf][3][j] + bv[3] : -__builtin_inff();
      float mx = fmaxf(fmaxf(x0, x1), fmaxf(x2, x3));
      #pragma unroll
      for (int d = 1; d <= 8; d <<= 1)     // reduce over the 16-lane half
        mx = fmaxf(mx, __shfl_xor(mx, d, 32));
      mx = fmaxf(mx, -1e30f);              // avoid -inf row max on dead chunks
      float s = __builtin_amdgcn_exp2f((x0 - mx) * LOG2E)
              + __builtin_amdgcn_exp2f((x1 - mx) * LOG2E)
              + __builtin_amdgcn_exp2f((x2 - mx) * LOG2E)
              + __builtin_amdgcn_exp2f((x3 - mx) * LOG2E);
      #pragma unroll
      for (int d = 1; d <= 8; d <<= 1)
        s += __shfl_xor(s, d, 32);
      int grow = rowBlk + wm * 64 + mf * 16 + j + half * 8;
      if (l16 == 0 && grow < kM) {
        pmax[(size_t)ch * kMP + grow] = mx;
        psum[(size_t)ch * kMP + grow] = s;
      }
    }
  }
}

// ---------------------------------------------------------------------------
// Kernel 2: true-class logit per row: t[m] = dot(e[m], W[y[m]]) + bias[y[m]]
// ---------------------------------------------------------------------------
__global__ __launch_bounds__(256)
void true_logit_kernel(const float* __restrict__ e, const float* __restrict__ w,
                       const float* __restrict__ bias, const int* __restrict__ labels,
                       float* __restrict__ t) {
  int m  = blockIdx.x;
  int bb = (m >= (kS - 1)) ? 1 : 0;
  int ss = m - bb * (kS - 1);
  int y  = labels[bb * kS + ss + 1];
  int ys = (y >= 0 && y < kV) ? y : 0;
  const float* er = e + (size_t)(bb * kS + ss) * kD;
  const float* wr = w + (size_t)ys * kD;
  float p = 0.0f;
  for (int d = threadIdx.x; d < kD; d += 256) p = fmaf(er[d], wr[d], p);
  #pragma unroll
  for (int d = 16; d >= 1; d >>= 1) p += __shfl_xor(p, d, 32);
  __shared__ float red[8];
  if ((threadIdx.x & 31) == 0) red[threadIdx.x >> 5] = p;
  __syncthreads();
  if (threadIdx.x == 0) {
    float q = 0.0f;
    #pragma unroll
    for (int i = 0; i < 8; ++i) q += red[i];
    t[m] = q + bias[ys];
  }
}

// ---------------------------------------------------------------------------
// Kernel 3: merge chunk partials -> lse per row; mean of (lse - true_logit).
// ---------------------------------------------------------------------------
__global__ __launch_bounds__(512)
void finalize_kernel(const float* __restrict__ pmax, const float* __restrict__ psum,
                     const float* __restrict__ t, const int* __restrict__ labels,
                     float* __restrict__ out) {
  float lsum = 0.0f;
  int   lcnt = 0;
  for (int m = threadIdx.x; m < kM; m += 512) {
    float M = -__builtin_inff(), S = 0.0f;
    for (int ch = 0; ch < NCH; ++ch) {
      float mi = pmax[(size_t)ch * kMP + m];
      float si = psum[(size_t)ch * kMP + m];
      if (mi > M) {
        S = S * __builtin_amdgcn_exp2f((M - mi) * LOG2E) + si;
        M = mi;
      } else {
        S += si * __builtin_amdgcn_exp2f((mi - M) * LOG2E);
      }
    }
    float lse = M + LN2 * __builtin_amdgcn_logf(S);
    int bb = (m >= (kS - 1)) ? 1 : 0;
    int ss = m - bb * (kS - 1);
    int y  = labels[bb * kS + ss + 1];
    if (y != IGNORE_INDEX) { lsum += lse - t[m]; lcnt += 1; }
  }
  __shared__ float ssum[512];
  __shared__ int   scnt[512];
  ssum[threadIdx.x] = lsum;
  scnt[threadIdx.x] = lcnt;
  __syncthreads();
  for (int off = 256; off > 0; off >>= 1) {
    if ((int)threadIdx.x < off) {
      ssum[threadIdx.x] += ssum[threadIdx.x + off];
      scnt[threadIdx.x] += scnt[threadIdx.x + off];
    }
    __syncthreads();
  }
  if (threadIdx.x == 0) out[0] = ssum[0] / (float)(scnt[0] > 0 ? scnt[0] : 1);
}

extern "C" void kernel_launch(void* const* d_in, const int* in_sizes, int n_in,
                              void* d_out, int out_size, void* d_ws, size_t ws_size,
                              hipStream_t stream) {
  (void)in_sizes; (void)n_in; (void)out_size; (void)ws_size;
  const float* e      = (const float*)d_in[0];
  const float* w      = (const float*)d_in[1];
  const float* bias   = (const float*)d_in[2];
  const int*   labels = (const int*)d_in[3];
  float* out = (float*)d_out;

  float* ws   = (float*)d_ws;
  float* pmax = ws;                               // NCH * kMP floats
  float* psum = ws + (size_t)NCH * kMP;           // NCH * kMP floats
  float* t    = ws + (size_t)2 * NCH * kMP;       // kMP floats  (~25.9 MB total)

  dim3 grid((kM + MT - 1) / MT, (kV + NT - 1) / NT);       // 16 x 197
  size_t shmem = (size_t)(MT + NT) * PITCH * sizeof(bf16_t); // 72 KB
  fused_logits_lse_kernel<<<grid, 512, shmem, stream>>>(e, w, bias, pmax, psum);
  true_logit_kernel<<<kM, 256, 0, stream>>>(e, w, bias, labels, t);
  finalize_kernel<<<1, 512, 0, stream>>>(pmax, psum, t, labels, out);
}
